// LevelAssociationNeuralNetworks_20023137534840
// MI455X (gfx1250) — compile-verified, matrix operand order checked by reading, not run on hardware
//
#include <hip/hip_runtime.h>
#include <hip/hip_bf16.h>

typedef __attribute__((ext_vector_type(16))) _Float16 v16h;
typedef __attribute__((ext_vector_type(8)))  float    v8f;

// ---- sizes (fixed by the reference) ----
#define NB   2048              // roots
#define CF   16                // fanout
#define DD   128               // input dim
#define HH   128               // hidden dim
#define EDD  8                 // edge dim
#define N1   (NB*CF)           // 32768
#define N2   (NB*CF*CF)        // 524288

// ---- gfx1250 async global->LDS copy (ASYNCcnt path), 16B per lane ----
__device__ __forceinline__ void async_copy16(const void* gp, const void* lp) {
  const unsigned lds = (unsigned)(size_t)lp;   // generic LDS addr low 32b = LDS offset
  asm volatile("global_load_async_to_lds_b128 %0, %1, off"
               :: "v"(lds), "v"(gp) : "memory");
}
__device__ __forceinline__ void wait_async() {
  asm volatile("s_wait_asynccnt 0x0" ::: "memory");
}

// Load a 16x32 f16 fragment (A layout; B uses the same layout on a
// transposed [N][K] tile). tile: row-major halves, 'stride' halves per row.
// Per CDNA5 ISA: lane m=L&15, half=L>>4; VGPR v holds K pair at
// Kb = (v>=4?16:0) + half*8 + (v&3)*2  -> two contiguous 16B chunks per lane.
__device__ __forceinline__ v16h load_frag(const _Float16* tile, int stride, int lane) {
  const int m = lane & 15, half = lane >> 4;
  union { v16h v; _Float16 h[16]; } f;
#pragma unroll
  for (int r = 0; r < 8; ++r) {
    const int kb = ((r >> 2) << 4) + (half << 3) + ((r & 3) << 1);
    const _Float16* p = tile + m * stride + kb;
    f.h[2 * r]     = p[0];
    f.h[2 * r + 1] = p[1];
  }
  return f.v;
}

// ---------------------------------------------------------------------------
// Kernel 1: fold feature-extractor into per-level weights.
//   WcT[l][j][i] = sum_k W_fe[i][k] * W_x[l][k][j]    (transposed, f16)
//   WhT/WgT    = transposed f16 copies of W_h/W_g
//   biasc[l][t][j] = b_fe @ W_x[l][:128,j] + W_x[l][128+t][j] + b_r[l][j]
// ---------------------------------------------------------------------------
__global__ void __launch_bounds__(256) prep_kernel(
    const float* __restrict__ Wfe, const float* __restrict__ bfe,
    const float* __restrict__ Wx,  const float* __restrict__ Wh,
    const float* __restrict__ Wg,  const float* __restrict__ br,
    _Float16* __restrict__ WcT, _Float16* __restrict__ WhT,
    _Float16* __restrict__ WgT, float* __restrict__ biasc) {
  const int l = blockIdx.x;
  const int tid = threadIdx.x;
  const float* WxL = Wx + l * 132 * 128;
  for (int idx = tid; idx < 128 * 128; idx += 256) {
    const int i = idx >> 7;     // K (input)
    const int j = idx & 127;    // N (output)
    float s = 0.f;
    for (int k = 0; k < 128; ++k) s += Wfe[i * 128 + k] * WxL[k * 128 + j];
    WcT[l * 16384 + j * 128 + i] = (_Float16)s;
    WhT[l * 16384 + j * 128 + i] = (_Float16)Wh[l * 16384 + i * 128 + j];
    WgT[l * 16384 + j * 128 + i] = (_Float16)Wg[l * 16384 + i * 128 + j];
  }
  for (int idx = tid; idx < 4 * 128; idx += 256) {
    const int t = idx >> 7, j = idx & 127;
    float s = 0.f;
    for (int k = 0; k < 128; ++k) s += bfe[k] * WxL[k * 128 + j];
    s += WxL[(128 + t) * 128 + j] + br[l * 128 + j];
    biasc[(l * 4 + t) * 128 + j] = s;
  }
}

// ---------------------------------------------------------------------------
// Kernel 2/4: h = tanh( X @ Wc  [+ Pooled @ Wh]  + biasc[t] )
// One block = 128 rows x 128 cols. 8 waves; wave w owns rows w*16..w*16+15.
// Weight / pooled tiles staged with global_load_async_to_lds_b128.
// ---------------------------------------------------------------------------
__global__ void __launch_bounds__(256) rnn_gemm_kernel(
    const float* __restrict__ X, const int* __restrict__ Tt,
    const _Float16* __restrict__ WcTg, const _Float16* __restrict__ WhTg,
    const _Float16* __restrict__ Pooled, const float* __restrict__ biasc,
    float* __restrict__ outF, _Float16* __restrict__ outH) {
  extern __shared__ char smem[];
  _Float16* aX  = (_Float16*)smem;              // [128][128] halves (reused for out)
  _Float16* bWc = aX + 16384;                   // [128 N][128 K] halves
  int*      tT  = (int*)(bWc + 16384);          // 128 ints
  _Float16* aP  = (_Float16*)(tT + 128);        // [128][128] (only if Pooled)
  _Float16* bWh = aP + 16384;

  const int tid  = threadIdx.x;
  const int row0 = blockIdx.x * 128;

  {  // stage WcT via async DMA to LDS (8 x 16B per thread)
#pragma unroll
    for (int i = 0; i < 8; ++i) {
      const int q = tid + i * 256;
      async_copy16((const uint4*)WcTg + q, (uint4*)bWc + q);
    }
  }
  if (Pooled) {
#pragma unroll
    for (int i = 0; i < 8; ++i) {
      const int q = tid + i * 256;
      async_copy16((const uint4*)(Pooled + (size_t)row0 * 128) + q, (uint4*)aP + q);
      async_copy16((const uint4*)WhTg + q, (uint4*)bWh + q);
    }
  }
  {  // stage X (f32 -> f16 conversion must go through VGPRs)
    const float4* Xg = (const float4*)(X + (size_t)row0 * 128);
#pragma unroll
    for (int i = 0; i < 16; ++i) {
      const int q = tid + i * 256;        // 4096 float4 = 16384 floats
      const float4 v = Xg[q];
      const int e = q * 4;
      aX[e + 0] = (_Float16)v.x; aX[e + 1] = (_Float16)v.y;
      aX[e + 2] = (_Float16)v.z; aX[e + 3] = (_Float16)v.w;
    }
  }
  if (tid < 128) tT[tid] = Tt[row0 + tid];
  wait_async();
  __syncthreads();

  const int w = tid >> 5, lane = tid & 31, half = lane >> 4, col = lane & 15;
  v8f acc[8];
#pragma unroll
  for (int n = 0; n < 8; ++n) acc[n] = (v8f){};

  const _Float16* aBase = aX + (w * 16) * 128;
#pragma unroll
  for (int kk = 0; kk < 4; ++kk) {
    const v16h a = load_frag(aBase + kk * 32, 128, lane);
    v16h b[8];
#pragma unroll
    for (int n = 0; n < 8; ++n) b[n] = load_frag(bWc + (n * 16) * 128 + kk * 32, 128, lane);
#pragma unroll
    for (int n = 0; n < 8; ++n)
      acc[n] = __builtin_amdgcn_wmma_f32_16x16x32_f16(false, a, false, b[n],
                                                      (short)0, acc[n], false, false);
  }
  if (Pooled) {
    const _Float16* aBase2 = aP + (w * 16) * 128;
#pragma unroll
    for (int kk = 0; kk < 4; ++kk) {
      const v16h a = load_frag(aBase2 + kk * 32, 128, lane);
      v16h b[8];
#pragma unroll
      for (int n = 0; n < 8; ++n) b[n] = load_frag(bWh + (n * 16) * 128 + kk * 32, 128, lane);
#pragma unroll
      for (int n = 0; n < 8; ++n)
        acc[n] = __builtin_amdgcn_wmma_f32_16x16x32_f16(false, a, false, b[n],
                                                        (short)0, acc[n], false, false);
    }
  }

  // epilogue: per-type bias + tanh
  if (outH) {
    __syncthreads();                      // done reading aX/aP -> reuse aX as out tile
#pragma unroll
    for (int n = 0; n < 8; ++n) {
#pragma unroll
      for (int r = 0; r < 8; ++r) {
        const int rl = w * 16 + half * 8 + r;
        const int c  = n * 16 + col;
        float v = acc[n][r] + biasc[tT[rl] * 128 + c];
        aX[rl * 128 + c] = (_Float16)tanhf(v);
      }
    }
    __syncthreads();
    const uint4* s = (const uint4*)aX;
    uint4* g = (uint4*)(outH + (size_t)row0 * 128);
#pragma unroll
    for (int i = 0; i < 8; ++i) g[tid + i * 256] = s[tid + i * 256];
  } else {
#pragma unroll
    for (int n = 0; n < 8; ++n) {
#pragma unroll
      for (int r = 0; r < 8; ++r) {
        const int rl = w * 16 + half * 8 + r;
        const int c  = n * 16 + col;
        float v = acc[n][r] + biasc[tT[rl] * 128 + c];
        outF[(size_t)(row0 + rl) * 128 + c] = tanhf(v);
      }
    }
  }
}

// ---------------------------------------------------------------------------
// Kernel 3: edge-gated graph conv + relu + max-pool readout.
// One wave per parent: S = A * sigmoid(E.We)  (16x16, pad K->32)
//   M = S @ childH (8 WMMAs), Hc = relu(M @ Wg + bg) (32 WMMAs), col-max pool.
// ---------------------------------------------------------------------------
__global__ void __launch_bounds__(256) gnn_kernel(
    const float* __restrict__ A, const float* __restrict__ E,
    const _Float16* __restrict__ childH, const float* __restrict__ We,
    const _Float16* __restrict__ WgTg, const float* __restrict__ bg,
    _Float16* __restrict__ pooledOut) {
  extern __shared__ char smem[];
  _Float16* sWg = (_Float16*)smem;  // [128 N][128 K] halves, shared by block
  const int tid = threadIdx.x, w = tid >> 5, lane = tid & 31;
  const int half = lane >> 4, col = lane & 15;
  _Float16* base = sWg + 16384 + w * (512 + 4096 + 2048);
  _Float16* sS  = base;              // [16][32]  S padded
  _Float16* chT = base + 512;        // [128 N][32 K] child_h^T padded
  _Float16* mT  = base + 512 + 4096; // [16][128] M tile (f16)

  {  // stage WgT once per block via async DMA
#pragma unroll
    for (int i = 0; i < 8; ++i) {
      const int q = tid + i * 256;
      async_copy16((const uint4*)WgTg + q, (uint4*)sWg + q);
    }
  }
  const long p = (long)blockIdx.x * 8 + w;
  float we[8];
#pragma unroll
  for (int i = 0; i < 8; ++i) we[i] = We[i];

  // gated adjacency S[c][k] = A[p,c,k] * sigmoid(E[p,c,k,:].We), K-pad zeros
  const float* Ep = E + p * (long)(CF * CF * EDD);
  const float* Ap = A + p * (long)(CF * CF);
#pragma unroll
  for (int i = 0; i < 8; ++i) {
    const int q = i * 32 + lane;          // 0..255
    const int c = q >> 4, k = q & 15;
    const float* e8 = Ep + q * 8;
    float d = 0.f;
#pragma unroll
    for (int j = 0; j < 8; ++j) d += e8[j] * we[j];
    const float g = 1.f / (1.f + __expf(-d));
    sS[c * 32 + k]      = (_Float16)(Ap[q] * g);
    sS[c * 32 + k + 16] = (_Float16)0.f;
  }
  // child hiddens -> transposed B layout [h][c]; vectorized b128 global loads
  {
    const uint4* chv = (const uint4*)(childH + p * (long)(CF * HH));  // 256 x 16B
#pragma unroll
    for (int i = 0; i < 8; ++i) {
      const int q4 = i * 32 + lane;       // 0..255
      union { uint4 u; _Float16 h[8]; } v;
      v.u = chv[q4];
      const int c  = q4 >> 4;             // child index
      const int h0 = (q4 * 8) & 127;      // first hidden dim of this chunk
#pragma unroll
      for (int j = 0; j < 8; ++j) chT[(h0 + j) * 32 + c] = v.h[j];
    }
    const uint4 z = {0, 0, 0, 0};
#pragma unroll
    for (int i = 0; i < 4; ++i) {
      const int h = i * 32 + lane;        // 0..127 : zero K-pad c=16..31
      *(uint4*)(chT + h * 32 + 16) = z;
      *(uint4*)(chT + h * 32 + 24) = z;
    }
  }
  wait_async();
  __syncthreads();

  // M = S @ childH
  v8f macc[8];
  {
    const v16h a = load_frag(sS, 32, lane);
    v16h b[8];
#pragma unroll
    for (int n = 0; n < 8; ++n) b[n] = load_frag(chT + n * 16 * 32, 32, lane);
#pragma unroll
    for (int n = 0; n < 8; ++n)
      macc[n] = __builtin_amdgcn_wmma_f32_16x16x32_f16(false, a, false, b[n],
                                                       (short)0, (v8f){}, false, false);
  }
#pragma unroll
  for (int n = 0; n < 8; ++n)
#pragma unroll
    for (int r = 0; r < 8; ++r)
      mT[(half * 8 + r) * 128 + n * 16 + col] = (_Float16)macc[n][r];
  __syncthreads();

  // Hc = relu(M @ Wg + bg)
  v8f acc[8];
#pragma unroll
  for (int n = 0; n < 8; ++n) acc[n] = (v8f){};
#pragma unroll
  for (int kk = 0; kk < 4; ++kk) {
    const v16h a = load_frag(mT + kk * 32, 128, lane);
    v16h b[8];
#pragma unroll
    for (int n = 0; n < 8; ++n) b[n] = load_frag(sWg + n * 16 * 128 + kk * 32, 128, lane);
#pragma unroll
    for (int n = 0; n < 8; ++n)
      acc[n] = __builtin_amdgcn_wmma_f32_16x16x32_f16(false, a, false, b[n],
                                                      (short)0, acc[n], false, false);
  }
  // relu + column-max over the 16 child rows (relu output >= 0 -> init 0)
  float* poolbuf = (float*)mT;  // reuse per-wave region
#pragma unroll
  for (int n = 0; n < 8; ++n) {
    const float bgv = bg[n * 16 + col];
    float pm = 0.f;
#pragma unroll
    for (int r = 0; r < 8; ++r) {
      float v = acc[n][r] + bgv;
      v = v > 0.f ? v : 0.f;
      pm = pm > v ? pm : v;
    }
    poolbuf[half * 128 + n * 16 + col] = pm;
  }
  __syncthreads();
  if (half == 0) {
#pragma unroll
    for (int n = 0; n < 8; ++n) {
      const int c = n * 16 + col;
      const float a0 = poolbuf[c], b0 = poolbuf[128 + c];
      pooledOut[p * 128 + c] = (_Float16)(a0 > b0 ? a0 : b0);
    }
  }
}

// ---------------------------------------------------------------------------
extern "C" void kernel_launch(void* const* d_in, const int* in_sizes, int n_in,
                              void* d_out, int out_size, void* d_ws, size_t ws_size,
                              hipStream_t stream) {
  const float* x0 = (const float*)d_in[0];
  const float* x1 = (const float*)d_in[1];
  const float* x2 = (const float*)d_in[2];
  const int*   t0 = (const int*)d_in[3];
  const int*   t1 = (const int*)d_in[4];
  const int*   t2 = (const int*)d_in[5];
  const float* A0 = (const float*)d_in[6];
  const float* A1 = (const float*)d_in[7];
  const float* E0 = (const float*)d_in[8];
  const float* E1 = (const float*)d_in[9];
  const float* Wfe = (const float*)d_in[10];
  const float* bfe = (const float*)d_in[11];
  const float* Wx  = (const float*)d_in[12];
  const float* Wh  = (const float*)d_in[13];
  const float* We  = (const float*)d_in[14];
  const float* Wg  = (const float*)d_in[15];
  const float* bg  = (const float*)d_in[16];
  const float* br  = (const float*)d_in[17];
  float* out = (float*)d_out;

  // workspace carve (all 16B aligned)
  _Float16* WcT = (_Float16*)d_ws;                       // 3*16384 h
  _Float16* WhT = WcT + 3 * 16384;
  _Float16* WgT = WhT + 3 * 16384;
  float* biasc  = (float*)(WgT + 3 * 16384);             // 3*4*128 f
  _Float16* pooled1 = (_Float16*)(biasc + 3 * 4 * 128);  // N1*128 h
  _Float16* pooled0 = pooled1 + (size_t)N1 * 128;        // NB*128 h
  _Float16* h1 = pooled0 + (size_t)NB * 128;             // N1*128 h
  _Float16* h2 = h1 + (size_t)N1 * 128;                  // N2*128 h

  const size_t shA  = 2 * 16384 * sizeof(_Float16) + 128 * sizeof(int);          // no pooled
  const size_t shAB = 4 * 16384 * sizeof(_Float16) + 128 * sizeof(int);          // with pooled
  const size_t shG  = 16384 * sizeof(_Float16) + 8 * (512 + 4096 + 2048) * sizeof(_Float16);

  hipLaunchKernelGGL(prep_kernel, dim3(3), dim3(256), 0, stream,
                     Wfe, bfe, Wx, Wh, Wg, br, WcT, WhT, WgT, biasc);

  // level 2: h2 = tanh(x2 @ Wc2 + bias2[t2])
  hipLaunchKernelGGL(rnn_gemm_kernel, dim3(N2 / 128), dim3(256), shA, stream,
                     x2, t2, WcT + 2 * 16384, (const _Float16*)nullptr,
                     (const _Float16*)nullptr, biasc + 2 * 4 * 128,
                     (float*)nullptr, h2);

  // level 1 readout + rnn
  hipLaunchKernelGGL(gnn_kernel, dim3(N1 / 8), dim3(256), shG, stream,
                     A1, E1, h2, We + 8, WgT + 16384, bg + 128, pooled1);
  hipLaunchKernelGGL(rnn_gemm_kernel, dim3(N1 / 128), dim3(256), shAB, stream,
                     x1, t1, WcT + 16384, WhT + 16384, pooled1,
                     biasc + 4 * 128, (float*)nullptr, h1);

  // level 0 readout + rnn -> d_out (f32)
  hipLaunchKernelGGL(gnn_kernel, dim3(NB / 8), dim3(256), shG, stream,
                     A0, E0, h1, We, WgT, bg, pooled0);
  hipLaunchKernelGGL(rnn_gemm_kernel, dim3(NB / 128), dim3(256), shAB, stream,
                     x0, t0, WcT, WhT, pooled0, biasc,
                     out, (_Float16*)nullptr);
}
